// OIMLoss_50637664420263
// MI455X (gfx1250) — matrix operand order repaired and checked by reading, not compiled, not run on hardware
//
#include <hip/hip_runtime.h>
#include <hip/hip_bf16.h>

typedef __attribute__((ext_vector_type(16))) __bf16 v16bf;
typedef __attribute__((ext_vector_type(8)))  __bf16 v8bf;
typedef __attribute__((ext_vector_type(8)))  float  v8f;

#define DIM      256
#define M_TOTAL  4096
#define N_LUT    5532
#define N_QUEUE  5000
#define N_TOTAL  (N_LUT + N_QUEUE)   // 10532

// ---------------------------------------------------------------------------
// Kernel 1: f32 -> bf16 hi/lo split planes in workspace.
//   A_hi[4096*256], A_lo[4096*256], W_hi[10532*256], W_lo[10532*256]
// W rows = concat(lut rows, queue rows)  (fuses the concat into the GEMM B).
// ---------------------------------------------------------------------------
__global__ void __launch_bounds__(256)
oim_convert_bf16x2(const float* __restrict__ inputs,
                   const float* __restrict__ lut,
                   const float* __restrict__ queue,
                   __bf16* __restrict__ a_hi, __bf16* __restrict__ a_lo,
                   __bf16* __restrict__ w_hi, __bf16* __restrict__ w_lo) {
    const int idx = blockIdx.x * blockDim.x + threadIdx.x;
    const int a_elems = M_TOTAL * DIM;
    const int w_elems = N_TOTAL * DIM;
    if (idx < a_elems) {
        float x   = inputs[idx];
        __bf16 h  = (__bf16)x;
        a_hi[idx] = h;
        a_lo[idx] = (__bf16)(x - (float)h);
    } else {
        int widx = idx - a_elems;
        if (widx < w_elems) {
            int row = widx >> 8;           // /DIM
            int k   = widx & (DIM - 1);
            float x = (row < N_LUT) ? lut[row * DIM + k]
                                    : queue[(row - N_LUT) * DIM + k];
            __bf16 h  = (__bf16)x;
            w_hi[widx] = h;
            w_lo[widx] = (__bf16)(x - (float)h);
        }
    }
}

// ---------------------------------------------------------------------------
// A-matrix 16x32 bf16 fragment (ISA 7.12.2 "16-bit A-Matrix 16x32"):
//   lane&15 = row M; lanes 0-15: K = kk+0..7 (v0..3) and kk+16..23 (v4..7);
//   lanes 16-31: K = kk+8..15 and kk+24..31.
// => per lane: two aligned 16-byte loads at elem offsets (kk + 8*half) and +16.
// ---------------------------------------------------------------------------
__device__ __forceinline__ v16bf load_a_frag(const __bf16* __restrict__ base,
                                             int row, int kk, int half) {
    const __bf16* p = base + row * DIM + kk + half * 8;
    v8bf c0 = *(const v8bf*)(p);        // K = +0..7   -> v0..v3
    v8bf c1 = *(const v8bf*)(p + 16);   // K = +16..23 -> v4..v7
    return __builtin_shufflevector(c0, c1,
        0, 1, 2, 3, 4, 5, 6, 7, 8, 9, 10, 11, 12, 13, 14, 15);
}

// ---------------------------------------------------------------------------
// B-matrix 32x16 bf16 fragment (ISA 7.12.4 B layout, dense):
//   lane&15 = column N (= row of row-major W, since B = W^T);
//   lanes 0-15 hold K = kk+0..7 (v0..3) and kk+16..23 (v4..7);
//   lanes 16-31 hold K = kk+8..15 and kk+24..31 (same order as A).
// => per lane: two aligned 16-byte loads at elem offsets (kk + 8*half) and +16.
// ---------------------------------------------------------------------------
__device__ __forceinline__ v16bf load_b_frag(const __bf16* __restrict__ base,
                                             int row, int kk, int half) {
    const __bf16* p = base + row * DIM + kk + half * 8;
    v8bf c0 = *(const v8bf*)(p);        // K = +0..7   -> v0..v3
    v8bf c1 = *(const v8bf*)(p + 16);   // K = +16..23 -> v4..v7
    return __builtin_shufflevector(c0, c1,
        0, 1, 2, 3, 4, 5, 6, 7, 8, 9, 10, 11, 12, 13, 14, 15);
}

// ---------------------------------------------------------------------------
// Kernel 2: C[4096,10532] = A @ W^T via bf16x2 split-precision WMMA.
// Block: 256 threads = 8 wave32, computes 128(M) x 128(N) of C.
// Wave w: m-offset = (w>>2)*64 (4 subtiles), n-offset = (w&3)*32 (2 subtiles)
// => 8 accumulators (v8f) per wave.
// Per 16x16 tile per K-step: 3 chained WMMAs (Alo*Bhi + Ahi*Blo + Ahi*Bhi)
// ~ fp32-accurate split-precision product.
// ---------------------------------------------------------------------------
__global__ void __launch_bounds__(256)
oim_gemm_wmma(const __bf16* __restrict__ a_hi, const __bf16* __restrict__ a_lo,
              const __bf16* __restrict__ w_hi, const __bf16* __restrict__ w_lo,
              float* __restrict__ out) {
    const int lane   = threadIdx.x & 31;
    const int wave   = threadIdx.x >> 5;
    const int m_base = blockIdx.y * 128 + (wave >> 2) * 64;   // always in range
    const int n_base = blockIdx.x * 128 + (wave & 3) * 32;    // may overrun N
    const int lrow   = lane & 15;
    const int half   = lane >> 4;                              // 0 or 1

    v8f acc[4][2];
    #pragma unroll
    for (int i = 0; i < 4; ++i)
        #pragma unroll
        for (int j = 0; j < 2; ++j)
            acc[i][j] = (v8f){0.f, 0.f, 0.f, 0.f, 0.f, 0.f, 0.f, 0.f};

    for (int kk = 0; kk < DIM; kk += 32) {
        v16bf bh[2], bl[2];
        #pragma unroll
        for (int j = 0; j < 2; ++j) {
            int nrow = n_base + 16 * j + lrow;
            if (nrow > N_TOTAL - 1) nrow = N_TOTAL - 1;  // clamp; masked at store
            bh[j] = load_b_frag(w_hi, nrow, kk, half);
            bl[j] = load_b_frag(w_lo, nrow, kk, half);
        }

        #pragma unroll
        for (int i = 0; i < 4; ++i) {
            const int mrow = m_base + 16 * i + lrow;
            v16bf ah = load_a_frag(a_hi, mrow, kk, half);
            v16bf al = load_a_frag(a_lo, mrow, kk, half);
            #pragma unroll
            for (int j = 0; j < 2; ++j) {
                acc[i][j] = __builtin_amdgcn_wmma_f32_16x16x32_bf16(
                    false, al, false, bh[j], (short)0, acc[i][j], false, false);
                acc[i][j] = __builtin_amdgcn_wmma_f32_16x16x32_bf16(
                    false, ah, false, bl[j], (short)0, acc[i][j], false, false);
                acc[i][j] = __builtin_amdgcn_wmma_f32_16x16x32_bf16(
                    false, ah, false, bh[j], (short)0, acc[i][j], false, false);
            }
        }
    }

    // C/D 16x16 f32 layout: VGPR r -> M = r + 8*(lane>=16), N = lane&15.
    const int mo = half << 3;
    const int nc = lane & 15;
    #pragma unroll
    for (int i = 0; i < 4; ++i) {
        #pragma unroll
        for (int j = 0; j < 2; ++j) {
            const int col = n_base + 16 * j + nc;
            if (col < N_TOTAL) {
                const int rbase = m_base + 16 * i + mo;
                #pragma unroll
                for (int r = 0; r < 8; ++r) {
                    out[(size_t)(rbase + r) * N_TOTAL + col] = acc[i][j][r];
                }
            }
        }
    }
}

// ---------------------------------------------------------------------------
extern "C" void kernel_launch(void* const* d_in, const int* in_sizes, int n_in,
                              void* d_out, int out_size, void* d_ws, size_t ws_size,
                              hipStream_t stream) {
    const float* inputs = (const float*)d_in[0];
    // d_in[1] = targets (int), d_in[2] = gt_flag (bool): unused by forward.
    const float* lut    = (const float*)d_in[3];
    const float* queue  = (const float*)d_in[4];
    float*       out    = (float*)d_out;

    __bf16* a_hi = (__bf16*)d_ws;
    __bf16* a_lo = a_hi + (size_t)M_TOTAL * DIM;
    __bf16* w_hi = a_lo + (size_t)M_TOTAL * DIM;
    __bf16* w_lo = w_hi + (size_t)N_TOTAL * DIM;
    // total ws use: 2*(4096+10532)*256*2 bytes ~= 14.3 MiB

    const int total  = (M_TOTAL + N_TOTAL) * DIM;
    const int blocks = (total + 255) / 256;
    oim_convert_bf16x2<<<blocks, 256, 0, stream>>>(inputs, lut, queue,
                                                   a_hi, a_lo, w_hi, w_lo);

    dim3 grid((N_TOTAL + 127) / 128, M_TOTAL / 128);   // (83, 32)
    oim_gemm_wmma<<<grid, 256, 0, stream>>>(a_hi, a_lo, w_hi, w_lo, out);
}
